// DeepGraphInfomax_2860448219919
// MI455X (gfx1250) — compile-verified, matrix operand order checked by reading, not run on hardware
//
#include <hip/hip_runtime.h>
#include <hip/hip_bf16.h>

// ---------------------------------------------------------------------------
// Problem constants (from reference)
// ---------------------------------------------------------------------------
#define NROWS 100000
#define DDIM  512
#define KCL   128
#define CTEMP 30.0f

typedef __attribute__((ext_vector_type(16))) __bf16 v16bf;
typedef __attribute__((ext_vector_type(8)))  __bf16 v8bf;
typedef __attribute__((ext_vector_type(8)))  float  v8f;

// Load a 16-element bf16 WMMA A/B fragment for one lane from an LDS row.
// Layout per CDNA5 ISA 7.12.2 (16-bit A 16x32): lane half 0 -> K {0..7,16..23},
// lane half 1 -> K {8..15,24..31}. Row stride keeps 16B alignment (40 elems).
__device__ inline v16bf load_frag(const __bf16* rowBase, int half) {
    union { v16bf v; v8bf h[2]; } u;
    const __bf16* p = rowBase + half * 8;
    u.h[0] = *(const v8bf*)(p);
    u.h[1] = *(const v8bf*)(p + 16);
    return u.v;
}

// Low 32 bits of a generic pointer to __shared__ = LDS byte offset
__device__ inline unsigned lds_off(const void* p) {
    return (unsigned)(size_t)p;
}

// ---------------------------------------------------------------------------
// Kernel 0: transpose + convert W_enc (DxD f32, row-major [k][n]) into
// Wt bf16 [n][k] so GEMM1 B-fragments are contiguous in K.
// ---------------------------------------------------------------------------
__global__ __launch_bounds__(256) void wconv_kernel(const float* __restrict__ W,
                                                    __bf16* __restrict__ Wt) {
    int idx = blockIdx.x * 256 + threadIdx.x;      // 512*512 = 262144 threads
    int k = idx >> 9;
    int n = idx & 511;
    Wt[(size_t)n * DDIM + k] = (__bf16)W[idx];
}

// ---------------------------------------------------------------------------
// Kernel 1: fused GEMM1 + row L2-normalize + mu scatter-add.
//   pos_z = normalize_rows(x @ W_enc)  via v_wmma_f32_16x16x32_bf16
// Block = 256 threads (8 waves), tile M=32 x N=512 (full width), K chunks of 32.
// B tiles stream global->LDS with GLOBAL_LOAD_ASYNC_TO_LDS_B128 (ASYNCcnt).
// Row sum-of-squares reduced wave-locally (shfl_xor) then across waves via LDS,
// accumulators scaled in registers, mu sums accumulated with f32 atomics.
// ---------------------------------------------------------------------------
__global__ __launch_bounds__(256) void gemm1_kernel(const float*  __restrict__ x,
                                                    const __bf16* __restrict__ Wt,
                                                    const int*    __restrict__ labels,
                                                    float* __restrict__ out_pos,
                                                    float* __restrict__ mu_sums,
                                                    float* __restrict__ mu_counts) {
    __shared__ __attribute__((aligned(16))) __bf16 sA[32][40];   // 2.5 KB
    __shared__ __attribute__((aligned(16))) __bf16 sB[512][40];  // 40 KB
    __shared__ float sNorm[32][8];
    __shared__ float sInv[32];
    __shared__ int   sLab[32];

    const int tid  = threadIdx.x;
    const int lane = tid & 31;
    const int wave = tid >> 5;
    const int half = lane >> 4;
    const int l    = lane & 15;
    const int rbase   = blockIdx.x * 32;
    const int wavecol = wave * 64;

    const unsigned sB_base = lds_off(&sB[0][0]);

    v8f acc[2][4] = {};

    for (int kbase = 0; kbase < DDIM; kbase += 32) {
        // A tile: 32x32 f32 -> bf16 (coalesced, 4 elems/thread)
#pragma unroll
        for (int i = 0; i < 4; ++i) {
            int e = tid + i * 256;
            int r = e >> 5, c = e & 31;
            sA[r][c] = (__bf16)x[(size_t)(rbase + r) * DDIM + kbase + c];
        }
        // B tile: 512(N) x 32(K) bf16 = 2048 x 16B transfers, async to LDS.
        // LDS row stride = 80B (64B data + 16B pad keeps fragment alignment).
#pragma unroll
        for (int i = 0; i < 8; ++i) {
            int e = tid + i * 256;
            int n = e >> 2, cp = e & 3;
            unsigned lds  = sB_base + (unsigned)(n * 80 + cp * 16);
            unsigned voff = (unsigned)(n * (DDIM * 2) + kbase * 2 + cp * 16);
            asm volatile("global_load_async_to_lds_b128 %0, %1, %2"
                         :: "v"(lds), "v"(voff), "s"(Wt) : "memory");
        }
        asm volatile("s_wait_asynccnt 0x0" ::: "memory");
        __syncthreads();

        v16bf afr0 = load_frag(&sA[l][0], half);
        v16bf afr1 = load_frag(&sA[16 + l][0], half);
#pragma unroll
        for (int t = 0; t < 4; ++t) {
            v16bf bfr = load_frag(&sB[wavecol + t * 16 + l][0], half);
            acc[0][t] = __builtin_amdgcn_wmma_f32_16x16x32_bf16(
                false, afr0, false, bfr, (short)0, acc[0][t], false, false);
            acc[1][t] = __builtin_amdgcn_wmma_f32_16x16x32_bf16(
                false, afr1, false, bfr, (short)0, acc[1][t], false, false);
        }
        __syncthreads();
    }

    // ---- Row sum-of-squares: wave-local partials, reduce over the 16 lanes
    // that share each row (within each half), then across waves via LDS.
    float ss[2][8] = {};
#pragma unroll
    for (int m = 0; m < 2; ++m)
#pragma unroll
        for (int t = 0; t < 4; ++t)
#pragma unroll
            for (int r = 0; r < 8; ++r) {
                float v = acc[m][t][r];
                ss[m][r] += v * v;
            }
#pragma unroll
    for (int m = 0; m < 2; ++m)
#pragma unroll
        for (int r = 0; r < 8; ++r) {
            float s = ss[m][r];
            s += __shfl_xor(s, 1);
            s += __shfl_xor(s, 2);
            s += __shfl_xor(s, 4);
            s += __shfl_xor(s, 8);
            if (l == 0) sNorm[m * 16 + half * 8 + r][wave] = s;
        }
    __syncthreads();
    if (tid < 32) {
        float n2 = 0.0f;
#pragma unroll
        for (int w = 0; w < 8; ++w) n2 += sNorm[tid][w];
        sInv[tid] = rsqrtf(n2);
        int lab = labels[rbase + tid];
        sLab[tid] = lab;
        atomicAdd(&mu_counts[lab], 1.0f);
    }
    __syncthreads();

    // ---- Scale accumulators, write pos_z, scatter-add into mu sums.
#pragma unroll
    for (int m = 0; m < 2; ++m)
#pragma unroll
        for (int t = 0; t < 4; ++t)
#pragma unroll
            for (int r = 0; r < 8; ++r) {
                int row = m * 16 + half * 8 + r;
                int col = wavecol + t * 16 + l;
                float v = acc[m][t][r] * sInv[row];
                out_pos[(size_t)(rbase + row) * DDIM + col] = v;
                atomicAdd(&mu_sums[(size_t)sLab[row] * DDIM + col], v);
            }
}

// ---------------------------------------------------------------------------
// Kernel 2: finalize mu = sums / max(count,1); emit f32 (output) + bf16 (GEMM2 B).
// ---------------------------------------------------------------------------
__global__ __launch_bounds__(256) void mu_final_kernel(const float* __restrict__ sums,
                                                       const float* __restrict__ counts,
                                                       float*  __restrict__ out_mu,
                                                       __bf16* __restrict__ mub) {
    int idx = blockIdx.x * 256 + threadIdx.x;      // 128*512 = 65536
    int k = idx >> 9;
    float c = counts[k];
    c = c > 1.0f ? c : 1.0f;
    float m = sums[idx] / c;
    out_mu[idx] = m;
    mub[idx]    = (__bf16)m;
}

// ---------------------------------------------------------------------------
// Kernel 3: GEMM2 dist = pos_z @ mu.T (N=128 full width), fused softmax(30*dist),
// argmax, and scatter-add for u. One block = 32 rows. B tile async to LDS.
// ---------------------------------------------------------------------------
__global__ __launch_bounds__(256) void gemm2_kernel(const float*  __restrict__ pos,
                                                    const __bf16* __restrict__ mub,
                                                    float* __restrict__ out_r,
                                                    float* __restrict__ out_dist,
                                                    float* __restrict__ u_sums,
                                                    float* __restrict__ u_counts) {
    __shared__ __attribute__((aligned(16))) __bf16 sA[32][40];    // 2.5 KB
    __shared__ __attribute__((aligned(16))) __bf16 sB[128][40];   // 10 KB
    __shared__ __attribute__((aligned(16))) float  sC[32][132];   // 16.9 KB
    __shared__ float sRed[32][8];
    __shared__ int   sArgC[32][8];
    __shared__ float sRowM[32];
    __shared__ int   sRowA[32];

    const int tid  = threadIdx.x;
    const int lane = tid & 31;
    const int wave = tid >> 5;
    const int half = lane >> 4;
    const int l    = lane & 15;
    const int rbase = blockIdx.x * 32;
    const int ncol  = wave * 16;                   // each wave owns 16 of 128 cols

    const unsigned sB_base = lds_off(&sB[0][0]);

    v8f acc[2] = {};

    for (int kbase = 0; kbase < DDIM; kbase += 32) {
#pragma unroll
        for (int i = 0; i < 4; ++i) {
            int e = tid + i * 256;
            int r = e >> 5, c = e & 31;
            sA[r][c] = (__bf16)pos[(size_t)(rbase + r) * DDIM + kbase + c];
        }
        // B tile: 128 x 32 bf16 = 512 x 16B transfers, async to LDS.
#pragma unroll
        for (int i = 0; i < 2; ++i) {
            int e = tid + i * 256;
            int n = e >> 2, cp = e & 3;
            unsigned lds  = sB_base + (unsigned)(n * 80 + cp * 16);
            unsigned voff = (unsigned)(n * (DDIM * 2) + kbase * 2 + cp * 16);
            asm volatile("global_load_async_to_lds_b128 %0, %1, %2"
                         :: "v"(lds), "v"(voff), "s"(mub) : "memory");
        }
        asm volatile("s_wait_asynccnt 0x0" ::: "memory");
        __syncthreads();

        v16bf a0 = load_frag(&sA[l][0], half);
        v16bf a1 = load_frag(&sA[16 + l][0], half);
        v16bf b0 = load_frag(&sB[ncol + l][0], half);
        acc[0] = __builtin_amdgcn_wmma_f32_16x16x32_bf16(
            false, a0, false, b0, (short)0, acc[0], false, false);
        acc[1] = __builtin_amdgcn_wmma_f32_16x16x32_bf16(
            false, a1, false, b0, (short)0, acc[1], false, false);
        __syncthreads();
    }

    // Dump dist tile to LDS for block-wide row softmax.
#pragma unroll
    for (int m = 0; m < 2; ++m)
#pragma unroll
        for (int r = 0; r < 8; ++r)
            sC[m * 16 + half * 8 + r][ncol + l] = acc[m][r];
    __syncthreads();

    const int row = tid & 31;
    const int ch  = tid >> 5;                      // 8 chunks of 16 columns

    // Phase 1: per-chunk max + argmax (first-occurrence tie-break like argmax).
    float mx = -3.0e38f; int arg = 0;
#pragma unroll
    for (int j = 0; j < 16; ++j) {
        float v = sC[row][ch * 16 + j];
        if (v > mx) { mx = v; arg = ch * 16 + j; }
    }
    sRed[row][ch] = mx;
    sArgC[row][ch] = arg;
    __syncthreads();
    if (tid < 32) {
        float rm = -3.0e38f; int ra = 0;
#pragma unroll
        for (int c = 0; c < 8; ++c) {
            float v = sRed[tid][c];
            if (v > rm) { rm = v; ra = sArgC[tid][c]; }
        }
        sRowM[tid] = rm;
        sRowA[tid] = ra;
    }
    __syncthreads();

    // Phase 2: write dist, exponentiate, partial sums.
    float rm = sRowM[row];
    float s = 0.0f;
#pragma unroll
    for (int j = 0; j < 16; ++j) {
        int col = ch * 16 + j;
        float d = sC[row][col];
        out_dist[(size_t)(rbase + row) * KCL + col] = d;
        float e = __expf(CTEMP * (d - rm));
        sC[row][col] = e;
        s += e;
    }
    sRed[row][ch] = s;
    __syncthreads();
    if (tid < 32) {
        float t = 0.0f;
#pragma unroll
        for (int c = 0; c < 8; ++c) t += sRed[tid][c];
        sRowM[tid] = 1.0f / t;
        atomicAdd(&u_counts[sRowA[tid]], 1.0f);
    }
    __syncthreads();

    // Phase 3: write r.
    float inv = sRowM[row];
#pragma unroll
    for (int j = 0; j < 16; ++j) {
        int col = ch * 16 + j;
        out_r[(size_t)(rbase + row) * KCL + col] = sC[row][col] * inv;
    }

    // Phase 4: scatter-add pos_z rows into u_sums by hard assignment.
#pragma unroll 4
    for (int i = 0; i < 64; ++i) {
        int e  = tid + i * 256;
        int rr = e >> 9, col = e & 511;
        int g  = rbase + rr;
        atomicAdd(&u_sums[(size_t)sRowA[rr] * DDIM + col],
                  pos[(size_t)g * DDIM + col]);
    }
}

// ---------------------------------------------------------------------------
// Kernel 4: finalize u.
// ---------------------------------------------------------------------------
__global__ __launch_bounds__(256) void u_final_kernel(const float* __restrict__ sums,
                                                      const float* __restrict__ counts,
                                                      float* __restrict__ out_u) {
    int idx = blockIdx.x * 256 + threadIdx.x;      // 65536
    int k = idx >> 9;
    float c = counts[k];
    c = c > 1.0f ? c : 1.0f;
    out_u[idx] = sums[idx] / c;
}

// ---------------------------------------------------------------------------
// Host launcher
// ---------------------------------------------------------------------------
extern "C" void kernel_launch(void* const* d_in, const int* in_sizes, int n_in,
                              void* d_out, int out_size, void* d_ws, size_t ws_size,
                              hipStream_t stream) {
    const float* x      = (const float*)d_in[0];   // [N, D]
    const float* W      = (const float*)d_in[1];   // [D, D]
    const int*   labels = (const int*)d_in[2];     // [N]

    // Outputs concatenated flat in return order: pos_z, mu, r, dist, u
    float* out_pos  = (float*)d_out;
    float* out_mu   = out_pos  + (size_t)NROWS * DDIM;
    float* out_r    = out_mu   + (size_t)KCL   * DDIM;
    float* out_dist = out_r    + (size_t)NROWS * KCL;
    float* out_u    = out_dist + (size_t)NROWS * KCL;

    // Workspace layout (~1.2 MB, 16B aligned throughout)
    char* p = (char*)d_ws;
    __bf16* Wt  = (__bf16*)p;  p += (size_t)DDIM * DDIM * 2;   // 512 KB
    __bf16* mub = (__bf16*)p;  p += (size_t)KCL  * DDIM * 2;   // 128 KB
    char* zbase = p;
    float* mu_sums   = (float*)p;  p += (size_t)KCL * DDIM * 4; // 256 KB
    float* u_sums    = (float*)p;  p += (size_t)KCL * DDIM * 4; // 256 KB
    float* mu_counts = (float*)p;  p += (size_t)KCL * 4;
    float* u_counts  = (float*)p;  p += (size_t)KCL * 4;
    size_t zbytes = (size_t)(p - zbase);

    hipMemsetAsync(zbase, 0, zbytes, stream);

    wconv_kernel  <<<(DDIM * DDIM) / 256, 256, 0, stream>>>(W, Wt);
    gemm1_kernel  <<<NROWS / 32,          256, 0, stream>>>(x, Wt, labels, out_pos,
                                                            mu_sums, mu_counts);
    mu_final_kernel<<<(KCL * DDIM) / 256, 256, 0, stream>>>(mu_sums, mu_counts,
                                                            out_mu, mub);
    gemm2_kernel  <<<NROWS / 32,          256, 0, stream>>>(out_pos, mub, out_r,
                                                            out_dist, u_sums, u_counts);
    u_final_kernel<<<(KCL * DDIM) / 256,  256, 0, stream>>>(u_sums, u_counts, out_u);
}
